// VectorQuantizer_90993177133715
// MI455X (gfx1250) — compile-verified, matrix operand order checked by reading, not run on hardware
//
#include <hip/hip_runtime.h>
#include <hip/hip_bf16.h>
#include <float.h>

typedef __attribute__((ext_vector_type(16))) _Float16 v16h;
typedef __attribute__((ext_vector_type(8)))  float    v8f;
typedef __attribute__((ext_vector_type(8)))  unsigned int v8u;

#define NUM_EMB 512
#define DIM     64
#define THW     16384            // 16*32*32 spatial positions per batch
#define NTOT    131072           // 8 * THW total vectors
#define QELEMS  8388608          // 8*64*16384 quantized elements
#define LOSS_OFF 8388608
#define IDX_OFF  8388609
#define ROWS_PER_BLOCK 128
#define NBLOCKS (NTOT / ROWS_PER_BLOCK)   // 1024

// workspace layout (in dwords):
//   [0, 16384)        packed fp16 B operands, dense [tc(64)][lane(32)][v(8)]
//   [16384, 16896)    codebook squared norms (512 f32)
//   [16896, 17920)    per-block loss partials (1024 f32)
#define WS_BPACK 0
#define WS_ENORM 16384
#define WS_PART  16896

// ---------------------------------------------------------------------------
// Prep: repack codebook into WMMA f16 B-operand lane layout + compute ||e||^2.
// B element (K,N) = embedding[N][K].  16-bit operand layout (ISA 7.12.2):
// lane L holds N = L&15; dword v, half h -> Kloc = (v<4 ? 2v+h : 16+2(v-4)+h)
// + 8*(L>>4); operand chunk c covers K = 32c + Kloc.
// ---------------------------------------------------------------------------
__global__ void vq_prep(const float* __restrict__ emb, float* __restrict__ ws) {
  int g = blockIdx.x * 256 + threadIdx.x;
  unsigned int* bp = (unsigned int*)ws + WS_BPACK;
  if (g < 2048) {                       // 64 tile-chunks x 32 lanes
    int lane  = g & 31, tc = g >> 5;
    int ct    = tc >> 1, chunk = tc & 1;
    int hi8   = (lane >> 4) << 3;
    int N     = (ct << 4) + (lane & 15);
    const float* er = emb + N * DIM + chunk * 32;
#pragma unroll
    for (int v = 0; v < 8; ++v) {
      int k0 = (v < 4) ? (2 * v + hi8) : (16 + 2 * (v - 4) + hi8);
      _Float16 lo = (_Float16)er[k0];
      _Float16 hi = (_Float16)er[k0 + 1];
      unsigned int u = (unsigned int)__builtin_bit_cast(unsigned short, lo)
                     | ((unsigned int)__builtin_bit_cast(unsigned short, hi) << 16);
      bp[tc * 256 + lane * 8 + v] = u;
    }
  } else if (g < 2048 + NUM_EMB) {      // codebook norms
    int k = g - 2048;
    const float* er = emb + k * DIM;
    float s = 0.f;
#pragma unroll
    for (int d = 0; d < DIM; ++d) s += er[d] * er[d];
    ws[WS_ENORM + k] = s;
  }
}

// ---------------------------------------------------------------------------
// Main: codebook split across the 8 waves of a block -- each wave keeps its
// 4 column tiles (64 codes) of B resident in registers for the whole block.
// All waves scan the same 16-row A tile (WGP$/L2-cached re-reads), so the
// inner loop is pure WMMA + min-updates with no LDS operand traffic.
// ---------------------------------------------------------------------------
__global__ void __launch_bounds__(256)
vq_main(const float* __restrict__ inp, const float* __restrict__ emb,
        float* ws, float* out) {
  __shared__ float lds_val[128];            // per-wave candidate values
  __shared__ int   lds_id[128];             // per-wave candidate indices
  __shared__ int   lds_idx[ROWS_PER_BLOCK]; // final argmin per row
  __shared__ float lds_red[256];

  const int t    = threadIdx.x;
  const int wave = t >> 5, lane = t & 31;
  const int nbase = blockIdx.x * ROWS_PER_BLOCK;   // contiguous rows, one batch
  const int b = nbase >> 14;                       // THW rows per batch
  const size_t bbase = (size_t)b * DIM * THW;

  // --- B operands: wave's 4 tiles x 2 K-chunks, registers for whole block ---
  const v8u* bpv = (const v8u*)((const unsigned int*)ws + WS_BPACK);
  v16h B0[4], B1[4];
  float en[4];
#pragma unroll
  for (int j = 0; j < 4; ++j) {
    int ct  = wave * 4 + j;                 // column tile (codes ct*16..+15)
    B0[j] = __builtin_bit_cast(v16h, bpv[(ct * 2    ) * 32 + lane]);
    B1[j] = __builtin_bit_cast(v16h, bpv[(ct * 2 + 1) * 32 + lane]);
    en[j] = ws[WS_ENORM + ct * 16 + (lane & 15)];
  }

  const int hi8 = (lane >> 4) << 3;

  for (int i = 0; i < 8; ++i) {             // 8 row-blocks of 16 rows
    const int nrow = nbase + i * 16;
    const int nloc = (nrow & (THW - 1)) + (lane & 15);

    // A operands: 16x64 fp32 -> two 16x32 fp16 tiles (coalesced 64B groups)
    v16h a0, a1;
#pragma unroll
    for (int v = 0; v < 8; ++v) {
      int k0 = (v < 4) ? (2 * v + hi8) : (16 + 2 * (v - 4) + hi8);
      a0[2 * v]     = (_Float16)inp[bbase + (size_t)(k0)      * THW + nloc];
      a0[2 * v + 1] = (_Float16)inp[bbase + (size_t)(k0 + 1)  * THW + nloc];
      a1[2 * v]     = (_Float16)inp[bbase + (size_t)(k0 + 32) * THW + nloc];
      a1[2 * v + 1] = (_Float16)inp[bbase + (size_t)(k0 + 33) * THW + nloc];
    }

    float best[8];
    int   bidx[8];
#pragma unroll
    for (int q = 0; q < 8; ++q) { best[q] = FLT_MAX; bidx[q] = 0; }

    // 4 independent accumulator chains -> scheduler can hide WMMA hazards
#pragma unroll
    for (int j = 0; j < 4; ++j) {
      v8f c = {};
      c = __builtin_amdgcn_wmma_f32_16x16x32_f16(false, a0, false, B0[j],
                                                 (short)0, c, false, false);
      c = __builtin_amdgcn_wmma_f32_16x16x32_f16(false, a1, false, B1[j],
                                                 (short)0, c, false, false);
      // rank by ||e||^2 - 2 x.e (||x||^2 constant per row -> same argmin)
      int N = ((wave * 4 + j) << 4) + (lane & 15);
#pragma unroll
      for (int q = 0; q < 8; ++q) {
        float d = en[j] - 2.0f * c[q];
        if (d < best[q]) { best[q] = d; bidx[q] = N; }
      }
    }

    // Butterfly min over the 16 lanes holding each row M
    // (C layout: lane L holds N-col = L&15, M = q + 8*(L>>4)); min-index ties.
#pragma unroll
    for (int off = 1; off < 16; off <<= 1) {
#pragma unroll
      for (int q = 0; q < 8; ++q) {
        float ov = __shfl_xor(best[q], off, 32);
        int   oi = __shfl_xor(bidx[q], off, 32);
        if (ov < best[q] || (ov == best[q] && oi < bidx[q])) {
          best[q] = ov; bidx[q] = oi;
        }
      }
    }
    if (lane == 0 || lane == 16) {
      int mo = (lane >> 4) << 3;
#pragma unroll
      for (int q = 0; q < 8; ++q) {
        lds_val[wave * 16 + mo + q] = best[q];
        lds_id [wave * 16 + mo + q] = bidx[q];
      }
    }
    __syncthreads();
    if (t < 16) {                            // merge the 8 waves' candidates
      float bv = lds_val[t];
      int   bi = lds_id[t];
#pragma unroll
      for (int w2 = 1; w2 < 8; ++w2) {
        float ov = lds_val[w2 * 16 + t];
        int   oi = lds_id [w2 * 16 + t];
        if (ov < bv || (ov == bv && oi < bi)) { bv = ov; bi = oi; }
      }
      lds_idx[i * 16 + t] = bi;
    }
    __syncthreads();
  }

  // Indices output (float-cast, flat (B,T,H,W) order == row order).
  if (t < ROWS_PER_BLOCK)
    out[IDX_OFF + nbase + t] = (float)lds_idx[t];

  // Quantized writeback (quantized_st == quantized in value) + loss partial.
  const int row = t & 127;
  const int dh  = (t >> 7) * 32;
  const int k   = lds_idx[row];
  const int nl  = (nbase & (THW - 1)) + row;
  float acc = 0.f;
#pragma unroll 8
  for (int d = dh; d < dh + 32; ++d) {
    float q = emb[k * DIM + d];
    float x = inp[bbase + (size_t)d * THW + nl];
    float diff = q - x;
    acc += diff * diff;
    out[bbase + (size_t)d * THW + nl] = q;
  }
  lds_red[t] = acc;
  __syncthreads();
  for (int s = 128; s > 0; s >>= 1) {
    if (t < s) lds_red[t] += lds_red[t + s];
    __syncthreads();
  }
  if (t == 0) ws[WS_PART + blockIdx.x] = lds_red[0];
}

// ---------------------------------------------------------------------------
// Finalize: deterministic fixed-order reduction of 1024 partials -> loss.
// loss = q_latent + 0.25*e_latent = 1.25 * mean((q-x)^2)
// ---------------------------------------------------------------------------
__global__ void vq_finalize(const float* __restrict__ ws, float* __restrict__ out) {
  __shared__ float red[256];
  int t = threadIdx.x;
  float s = 0.f;
#pragma unroll
  for (int i = 0; i < 4; ++i) s += ws[WS_PART + t + i * 256];
  red[t] = s;
  __syncthreads();
  for (int k = 128; k > 0; k >>= 1) {
    if (t < k) red[t] += red[t + k];
    __syncthreads();
  }
  if (t == 0) out[LOSS_OFF] = 1.25f * red[0] / (float)QELEMS;
}

extern "C" void kernel_launch(void* const* d_in, const int* in_sizes, int n_in,
                              void* d_out, int out_size, void* d_ws, size_t ws_size,
                              hipStream_t stream) {
  const float* inp = (const float*)d_in[0];   // (8, 64, 16, 32, 32) f32
  const float* emb = (const float*)d_in[1];   // (512, 64) f32
  float* out = (float*)d_out;                 // [quantized | loss | indices]
  float* ws  = (float*)d_ws;

  vq_prep    <<<10,      256, 0, stream>>>(emb, ws);
  vq_main    <<<NBLOCKS, 256, 0, stream>>>(inp, emb, ws, out);
  vq_finalize<<<1,       256, 0, stream>>>(ws, out);
}
